// LinearAttention_48825188221478
// MI455X (gfx1250) — compile-verified, matrix operand order checked by reading, not run on hardware
//
#include <hip/hip_runtime.h>
#include <hip/hip_bf16.h>

typedef __attribute__((ext_vector_type(16))) __bf16 bf16x16;
typedef __attribute__((ext_vector_type(8)))  __bf16 bf16x8;
typedef __attribute__((ext_vector_type(8)))  float  f32x8;

#define B_   16
#define C_   256
#define H_   64
#define W_   64
#define HIDP 704   // 682 padded to 22*32
#define HID  682

// ---------------- WMMA helpers ----------------

__device__ __forceinline__ f32x8 wmma_bf16(bf16x16 a, bf16x16 b, f32x8 c) {
  return __builtin_amdgcn_wmma_f32_16x16x32_bf16(false, a, false, b, (short)0, c, false, false);
}

// 16x32 (bf16) A/B fragment from row-major [row][k] storage.
// Per ISA layout: lanes 0-15 row=lane, elems 0..7 -> K = k0+half*8+j, elems 8..15 -> K = k0+16+half*8+j.
__device__ __forceinline__ bf16x16 load_frag(const __bf16* base, int row0, int rowStride, int k0, int lane) {
  const __bf16* p = base + (size_t)(row0 + (lane & 15)) * rowStride + k0 + ((lane >> 4) << 3);
  union { bf16x16 v; bf16x8 h[2]; } u;
  u.h[0] = *(const bf16x8*)(p);
  u.h[1] = *(const bf16x8*)(p + 16);
  return u.v;
}

__device__ __forceinline__ float sigm(float x) { return 1.f / (1.f + __expf(-x)); }

// ---------------- minGRU GEMM: A(64x256) x W^T(512x256) -> a,b bf16 in LDS raw[64][512] ----------------
// Wave computes paired (h,z) tiles so gate math stays in registers.
__device__ __forceinline__ void mingru_gemm(const __bf16* A, const __bf16* Wg, __bf16* raw, int tid) {
  const int wv = tid >> 5, lane = tid & 31;
  const int half = lane >> 4, nl = lane & 15;
#pragma unroll 1
  for (int i = 0; i < 8; ++i) {
    const int pr = (wv << 3) | i;
    const int mt = pr >> 4;      // 0..3  (M tile)
    const int ct = pr & 15;      // 0..15 (channel tile)
    f32x8 acch = {}, accz = {};
#pragma unroll
    for (int kt = 0; kt < 8; ++kt) {
      bf16x16 af = load_frag(A,  mt * 16,        C_, kt * 32, lane);
      bf16x16 bh = load_frag(Wg, ct * 16,        C_, kt * 32, lane);
      bf16x16 bz = load_frag(Wg, 256 + ct * 16,  C_, kt * 32, lane);
      acch = wmma_bf16(af, bh, acch);
      accz = wmma_bf16(af, bz, accz);
    }
    const int cch = ct * 16 + nl;
#pragma unroll
    for (int j = 0; j < 8; ++j) {
      const int m = mt * 16 + j + (half << 3);
      float z  = sigm(accz[j]);
      float hr = acch[j];
      float g  = (hr >= 0.f) ? (hr + 0.5f) : sigm(hr);
      raw[m * 512 + cch]       = (__bf16)(1.f - z);   // a
      raw[m * 512 + 256 + cch] = (__bf16)(z * g);     // b
    }
  }
}

// ---------------- sequential scan over L=64, residual, emit to global or LDS ----------------
__device__ __forceinline__ void scan_seq(const __bf16* xin, __bf16* raw, __bf16* outg,
                                         bool rev, bool toLds, int tid) {
  const int c = tid;  // one channel per thread
  float h = 0.f;
#pragma unroll 4
  for (int s = 0; s < 64; ++s) {
    const int m = rev ? (63 - s) : s;
    float a  = (float)raw[m * 512 + c];
    float bb = (float)raw[m * 512 + 256 + c];
    h = a * h + bb;
    float xv = (float)xin[m * C_ + c] + h;
    if (toLds) raw[m * 512 + c] = (__bf16)xv;              // becomes A for projection
    else       outg[(size_t)m * (H_ * C_) + c] = (__bf16)xv;
  }
}

// ---------------- projection accumulate: p += cat_slice(64x256) @ projW_slice^T ----------------
__device__ __forceinline__ void proj_accum(const __bf16* raw, const __bf16* projW, int phase,
                                           f32x8* accp, int tid) {
  const int wv = tid >> 5, lane = tid & 31;
  const int mt = wv >> 1, ntb = (wv & 1) * 8;
#pragma unroll 1
  for (int t = 0; t < 8; ++t) {
    f32x8 acc = accp[t];
#pragma unroll
    for (int kt = 0; kt < 8; ++kt) {
      bf16x16 af  = load_frag(raw,   mt * 16,        512,  kt * 32,               lane);
      bf16x16 bf2 = load_frag(projW, (ntb + t) * 16, 1024, phase * 256 + kt * 32, lane);
      acc = wmma_bf16(af, bf2, acc);
    }
    accp[t] = acc;
  }
}

// ================= Kernel 1: row pass (transpose + RMSNorm + minGRU L/R) =================
__global__ __launch_bounds__(256) void k_rows(const float* __restrict__ x,
                                              const float* __restrict__ attn_w,
                                              const __bf16* __restrict__ wl,
                                              const __bf16* __restrict__ wr,
                                              __bf16* __restrict__ tN,
                                              __bf16* __restrict__ xl,
                                              __bf16* __restrict__ xr) {
  __shared__ __align__(16) unsigned char smem[65536];
  __bf16* raw = (__bf16*)smem;
  const int tid  = threadIdx.x;
  const int seq  = blockIdx.x;         // b*64 + hrow
  const int b    = seq >> 6, hrow = seq & 63;

  { // phase 0: gather x (B,C,H,W), RMSNorm over C, write bf16 tN (seq-major)
    const int w = tid & 63, q = tid >> 6;
    const float* xp = x + ((size_t)b * C_ * H_ + (size_t)hrow) * W_ + w;  // +c*H*W
    float v[64];
    float ss = 0.f;
#pragma unroll
    for (int i = 0; i < 64; ++i) {
      float t = xp[(size_t)(q * 64 + i) * (H_ * W_)];
      v[i] = t; ss += t * t;
    }
    float* part = (float*)smem;
    part[q * 64 + w] = ss;
    __syncthreads();
    float tot  = part[w] + part[64 + w] + part[128 + w] + part[192 + w];
    float rstd = rsqrtf(tot * (1.f / C_) + 1e-6f);
    __bf16* tn = tN + (size_t)seq * (W_ * C_) + (size_t)w * C_ + q * 64;
    const float* aw = attn_w + q * 64;
#pragma unroll
    for (int i = 0; i < 64; ++i) tn[i] = (__bf16)(v[i] * rstd * aw[i]);
    __threadfence_block();
  }
  __syncthreads();

  const __bf16* Aseq = tN + (size_t)seq * (W_ * C_);
  __bf16* outL = xl + (size_t)b * (W_ * H_ * C_) + (size_t)hrow * C_;  // (B,W,H,C), stride m = H*C
  __bf16* outR = xr + (size_t)b * (W_ * H_ * C_) + (size_t)hrow * C_;

  mingru_gemm(Aseq, wl, raw, tid);
  __syncthreads();
  scan_seq(Aseq, raw, outL, false, false, tid);
  __syncthreads();
  mingru_gemm(Aseq, wr, raw, tid);
  __syncthreads();
  scan_seq(Aseq, raw, outR, true, false, tid);
}

// ================= Kernel 2: column pass (4 minGRUs + fused projection) =================
__global__ __launch_bounds__(256) void k_cols(const __bf16* __restrict__ xl,
                                              const __bf16* __restrict__ xr,
                                              const __bf16* __restrict__ wt,
                                              const __bf16* __restrict__ wb,
                                              const __bf16* __restrict__ projW,
                                              float* __restrict__ p) {
  __shared__ __align__(16) unsigned char smem[65536];
  __bf16* raw = (__bf16*)smem;
  const int tid = threadIdx.x;
  const int seq = blockIdx.x;  // b*64 + w

  f32x8 accp[8];
#pragma unroll
  for (int t = 0; t < 8; ++t) { f32x8 z = {}; accp[t] = z; }

  const __bf16* srcs[2] = { xl + (size_t)seq * (H_ * C_), xr + (size_t)seq * (H_ * C_) };
#pragma unroll 1
  for (int si = 0; si < 2; ++si) {
#pragma unroll 1
    for (int dir = 0; dir < 2; ++dir) {
      mingru_gemm(srcs[si], (dir == 0) ? wt : wb, raw, tid);
      __syncthreads();
      scan_seq(srcs[si], raw, nullptr, dir == 1, true, tid);
      __syncthreads();
      proj_accum(raw, projW, si * 2 + dir, accp, tid);
      __syncthreads();
    }
  }

  // write p (B,W,H,C): [(seq*64 + m)*256 + n]
  const int wv = tid >> 5, lane = tid & 31;
  const int half = lane >> 4, nl = lane & 15;
  const int mt = wv >> 1, ntb = (wv & 1) * 8;
#pragma unroll
  for (int t = 0; t < 8; ++t) {
    const int nch = (ntb + t) * 16 + nl;
#pragma unroll
    for (int j = 0; j < 8; ++j) {
      const int m = mt * 16 + j + (half << 3);
      p[((size_t)seq * 64 + m) * 256 + nch] = accp[t][j];
    }
  }
}

// ================= Kernel 3: residual + channel-RMS + MLP =================
__global__ __launch_bounds__(256) void k_mlp(const float* __restrict__ x,
                                             const float* __restrict__ p,
                                             const float* __restrict__ norm_w,
                                             const __bf16* __restrict__ m1,
                                             const __bf16* __restrict__ m2,
                                             float* __restrict__ out) {
  __shared__ __align__(16) unsigned char smem[61440];
  __bf16* ybf = (__bf16*)smem;                 // [32][256] bf16
  __bf16* h1  = (__bf16*)(smem + 16384);       // [32][704] bf16
  float*  tmp = (float*)(smem + 16384);        // [32][256] f32 (overlaps h1, dead before GEMM1)
  float*  part = (float*)smem;                 // [8][32] scratch (overlaps ybf, dead before y write)

  const int tid = threadIdx.x;
  const int blk = blockIdx.x;
  const int whalf = blk & 1, bh = blk >> 1;
  const int b = bh >> 6, hrow = bh & 63;
  const int w0 = whalf * 32;

  { // stage s = xi + p, channel-RMS, write y bf16
    const int pix = tid & 31, q = tid >> 5;    // wave q owns channels q*32..q*32+31
    const float* xp = x + ((size_t)b * C_ * H_ + (size_t)hrow) * W_ + w0 + pix;        // +c*H*W
    const float* pp = p + (((size_t)b * 64 + (w0 + pix)) * 64 + hrow) * 256;           // +c
    float ss = 0.f;
#pragma unroll
    for (int i = 0; i < 32; ++i) {
      const int c = q * 32 + i;
      float v = xp[(size_t)c * (H_ * W_)] + pp[c];
      tmp[pix * 256 + c] = v;
      ss += v * v;
    }
    part[q * 32 + pix] = ss;
    __syncthreads();
    float tot = 0.f;
#pragma unroll
    for (int qq = 0; qq < 8; ++qq) tot += part[qq * 32 + pix];
    float rstd = rsqrtf(tot * (1.f / 256.f) + 1e-6f);
    __syncthreads();  // all 'part' reads done before ybf overwrite
#pragma unroll
    for (int i = 0; i < 32; ++i) {
      const int c = q * 32 + i;
      ybf[pix * 256 + c] = (__bf16)(tmp[pix * 256 + c] * rstd * norm_w[c]);
    }
  }
  __syncthreads();

  const int wv = tid >> 5, lane = tid & 31;
  const int half = lane >> 4, nl = lane & 15;

  // GEMM1: (32x256) @ m1^T(704x256) -> silu -> h1 bf16; 2x44 tiles, 11 per wave
#pragma unroll 1
  for (int t = 0; t < 11; ++t) {
    const int tt = wv * 11 + t;
    const int mt = tt / 44, nt = tt % 44;
    f32x8 acc = {};
#pragma unroll
    for (int kt = 0; kt < 8; ++kt) {
      bf16x16 af  = load_frag(ybf, mt * 16, 256, kt * 32, lane);
      bf16x16 bf2 = load_frag(m1,  nt * 16, 256, kt * 32, lane);
      acc = wmma_bf16(af, bf2, acc);
    }
    const int n = nt * 16 + nl;
#pragma unroll
    for (int j = 0; j < 8; ++j) {
      const int m = mt * 16 + j + (half << 3);
      float v = acc[j];
      h1[m * HIDP + n] = (__bf16)(v * sigm(v));   // silu
    }
  }
  __syncthreads();

  // GEMM2: (32x704) @ m2^T(256x704) -> out = y + m; 2x16 tiles, 4 per wave
#pragma unroll 1
  for (int t = 0; t < 4; ++t) {
    const int tt = wv * 4 + t;
    const int mt = tt >> 4, nt = tt & 15;
    f32x8 acc = {};
#pragma unroll 2
    for (int kt = 0; kt < 22; ++kt) {
      bf16x16 af  = load_frag(h1, mt * 16, HIDP, kt * 32, lane);
      bf16x16 bf2 = load_frag(m2, nt * 16, HIDP, kt * 32, lane);
      acc = wmma_bf16(af, bf2, acc);
    }
    const int c = nt * 16 + nl;
#pragma unroll
    for (int j = 0; j < 8; ++j) {
      const int m = mt * 16 + j + (half << 3);
      float yv = (float)ybf[m * 256 + c];
      out[(((size_t)b * C_ + c) * H_ + hrow) * W_ + w0 + m] = yv + acc[j];
    }
  }
}

// ================= weight conversion kernels =================
__global__ void cvt_bf16(const float* __restrict__ in, __bf16* __restrict__ o, int n) {
  int i = blockIdx.x * 256 + threadIdx.x;
  if (i < n) o[i] = (__bf16)in[i];
}
__global__ void cvt_pad_rows(const float* __restrict__ in, __bf16* __restrict__ o,
                             int rows_in, int rows_out, int cols) {
  int i = blockIdx.x * 256 + threadIdx.x;
  if (i < rows_out * cols) {
    int r = i / cols;
    o[i] = (r < rows_in) ? (__bf16)in[i] : (__bf16)0.f;
  }
}
__global__ void cvt_pad_cols(const float* __restrict__ in, __bf16* __restrict__ o,
                             int rows, int cols_in, int cols_out) {
  int i = blockIdx.x * 256 + threadIdx.x;
  if (i < rows * cols_out) {
    int r = i / cols_out, c = i % cols_out;
    o[i] = (c < cols_in) ? (__bf16)in[r * cols_in + c] : (__bf16)0.f;
  }
}

// ================= host launcher =================
extern "C" void kernel_launch(void* const* d_in, const int* in_sizes, int n_in,
                              void* d_out, int out_size, void* d_ws, size_t ws_size,
                              hipStream_t stream) {
  (void)in_sizes; (void)n_in; (void)out_size; (void)ws_size;
  const float* x      = (const float*)d_in[0];
  const float* attn_w = (const float*)d_in[1];
  const float* norm_w = (const float*)d_in[2];
  const float* w_l    = (const float*)d_in[3];
  const float* w_r    = (const float*)d_in[4];
  const float* w_t    = (const float*)d_in[5];
  const float* w_b    = (const float*)d_in[6];
  const float* proj_w = (const float*)d_in[7];
  const float* mlp_w1 = (const float*)d_in[8];
  const float* mlp_w2 = (const float*)d_in[9];
  float* out = (float*)d_out;

  unsigned char* ws = (unsigned char*)d_ws;
  size_t o = 0;
  const size_t ACT = (size_t)1024 * 64 * 256;        // 16.7M elems
  __bf16* tN = (__bf16*)(ws + o); o += ACT * 2;
  __bf16* xl = (__bf16*)(ws + o); o += ACT * 2;
  __bf16* xr = (__bf16*)(ws + o); o += ACT * 2;
  float*  pB = (float*)(ws + o);  o += ACT * 4;
  __bf16* wl = (__bf16*)(ws + o); o += (size_t)512 * 256 * 2;
  __bf16* wr = (__bf16*)(ws + o); o += (size_t)512 * 256 * 2;
  __bf16* wt = (__bf16*)(ws + o); o += (size_t)512 * 256 * 2;
  __bf16* wb = (__bf16*)(ws + o); o += (size_t)512 * 256 * 2;
  __bf16* pj = (__bf16*)(ws + o); o += (size_t)256 * 1024 * 2;
  __bf16* m1 = (__bf16*)(ws + o); o += (size_t)HIDP * 256 * 2;
  __bf16* m2 = (__bf16*)(ws + o); o += (size_t)256 * HIDP * 2;

  int n = 512 * 256;
  cvt_bf16<<<(n + 255) / 256, 256, 0, stream>>>(w_l, wl, n);
  cvt_bf16<<<(n + 255) / 256, 256, 0, stream>>>(w_r, wr, n);
  cvt_bf16<<<(n + 255) / 256, 256, 0, stream>>>(w_t, wt, n);
  cvt_bf16<<<(n + 255) / 256, 256, 0, stream>>>(w_b, wb, n);
  n = 256 * 1024;
  cvt_bf16<<<(n + 255) / 256, 256, 0, stream>>>(proj_w, pj, n);
  n = HIDP * 256;
  cvt_pad_rows<<<(n + 255) / 256, 256, 0, stream>>>(mlp_w1, m1, HID, HIDP, 256);
  n = 256 * HIDP;
  cvt_pad_cols<<<(n + 255) / 256, 256, 0, stream>>>(mlp_w2, m2, 256, HID, HIDP);

  k_rows<<<1024, 256, 0, stream>>>(x, attn_w, wl, wr, tN, xl, xr);
  k_cols<<<1024, 256, 0, stream>>>(xl, xr, wt, wb, pj, pB);
  k_mlp<<<2048, 256, 0, stream>>>(x, pB, norm_w, m1, m2, out);
}
